// LVC_9929964389080
// MI455X (gfx1250) — matrix operand order repaired and checked by reading, not compiled, not run on hardware
//
#include <hip/hip_runtime.h>
#include <hip/hip_bf16.h>

typedef __attribute__((ext_vector_type(16))) __bf16 v16bf;
typedef __attribute__((ext_vector_type(8)))  float  v8f;
typedef __attribute__((ext_vector_type(4)))  unsigned int u32x4;
typedef __attribute__((ext_vector_type(4)))  float  f32x4;
typedef __attribute__((ext_vector_type(8)))  int    i32x8;
typedef __attribute__((ext_vector_type(4)))  int    i32x4;

#define C_CH   256
#define NCODE  64
#define HW     4096
#define NB     16
#define XS     264   // LDS row stride (ushorts) for x / xp [pixel][channel]
#define DSTR   68    // dist row stride (floats)
#define ATS    72    // row stride (ushorts) for assign^T [code][pixel] and xp^T [channel][pixel]
#define SSTR   68    // TDM staging row stride (floats): 64 data + 4 pad dwords

__device__ __forceinline__ unsigned short f2bf_u(float f) {
    unsigned u = __builtin_bit_cast(unsigned, f);
    return (unsigned short)((u + 0x7FFFu + ((u >> 16) & 1u)) >> 16);
}

// A-matrix 16x32 bf16 fragment (ISA 7.12.2): lane-half kh: elems0-7 = K[32kt+8kh .. +7],
// elems8-15 = K[32kt+8kh+16 .. +23]
__device__ __forceinline__ v16bf fragA(const unsigned short* row, int kt, int kh) {
    int kb = kt * 32 + kh * 8;
    union { u32x4 q[2]; v16bf v; } u;
    u.q[0] = *reinterpret_cast<const u32x4*>(row + kb);
    u.q[1] = *reinterpret_cast<const u32x4*>(row + kb + 16);
    return u.v;
}
// B-matrix 32x16 bf16 fragment: lane holds column n=lane&15; elems0-7 = K[32kt+8kh .. +7], elems8-15 = K[32kt+8kh+16 .. +23]
__device__ __forceinline__ v16bf fragB(const unsigned short* row, int kt, int kh) {
    int kb = kt * 32 + kh * 8;
    union { u32x4 q[2]; v16bf v; } u;
    u.q[0] = *reinterpret_cast<const u32x4*>(row + kb);
    u.q[1] = *reinterpret_cast<const u32x4*>(row + kb + 16);
    return u.v;
}

__device__ __forceinline__ v8f wmma_bf16(v16bf a, v16bf b, v8f c) {
    return __builtin_amdgcn_wmma_f32_16x16x32_bf16(false, a, false, b, (short)0, c, false, false);
}

// TDM: async DMA of one 64x256 f32 tile (row stride HW) into LDS, 68-dword padded rows.
// D# per CDNA5 ISA §8.3/8.4; groups 2/3 zero (2-D tensor).
__device__ __forceinline__ void tdm_load_tile(const float* gsrc, unsigned ldsOff) {
    unsigned long long ga = (unsigned long long)(uintptr_t)gsrc;
    u32x4 g0;
    g0[0] = 1u;                                            // count=1, user desc, no gather
    g0[1] = ldsOff;                                        // lds_addr (bytes)
    g0[2] = (unsigned)ga;                                  // global_addr[31:0]
    g0[3] = (unsigned)((ga >> 32) & 0x01FFFFFFu) | (2u << 30); // addr[56:32] | type=2
    i32x8 g1;
    g1[0] = (2 << 16) | (1 << 20) | (5 << 22) | (3 << 25); // data_size=4B, pad_en, every 64dw pad 4dw
    g1[1] = (int)(64u << 16);                              // tensor_dim0 = 64 (low16 in [31:16])
    g1[2] = (int)(256u << 16);                             // tensor_dim0 hi=0 | tensor_dim1 = 256
    g1[3] = (int)(64u << 16);                              // tensor_dim1 hi=0 | tile_dim0 = 64
    g1[4] = 256;                                           // tile_dim1 = 256, tile_dim2 = 0
    g1[5] = HW;                                            // tensor_dim0_stride = 4096 elements
    g1[6] = 0;
    g1[7] = 0;
    i32x4 gz = (i32x4){0, 0, 0, 0};
#if __has_include(<hip/amd_detail/amd_gfx1250_TDM.h>)
    i32x8 gz8 = (i32x8){0, 0, 0, 0, 0, 0, 0, 0};
    __builtin_amdgcn_tensor_load_to_lds(g0, g1, gz, gz, gz8, 0);
#else
    __builtin_amdgcn_tensor_load_to_lds(g0, g1, gz, gz, 0);
#endif
}

// ---------------- prep: bf16 conversions, c2, smooth, zero accumulators ----------------
__global__ __launch_bounds__(256) void prep_kernel(
    const float* __restrict__ conv_w, const float* __restrict__ codebook,
    const float* __restrict__ log_smooth,
    unsigned short* __restrict__ wb, unsigned short* __restrict__ cb,
    float* __restrict__ c2, float* __restrict__ smooth,
    float* __restrict__ eAcc, float* __restrict__ aSum)
{
    int t = blockIdx.x * 256 + threadIdx.x;            // grid = 1024 * 256 = 262144
    if (t < C_CH * C_CH) wb[t] = f2bf_u(conv_w[t]);
    if (t < NCODE * C_CH) cb[t] = f2bf_u(codebook[t]);
    if (t < NCODE) {
        float s = 0.f;
        for (int c = 0; c < C_CH; ++c) { float v = codebook[t * C_CH + c]; s += v * v; }
        c2[t] = s;
        smooth[t] = __expf(log_smooth[t]);
    }
    if (t < NB * NCODE * C_CH) eAcc[t] = 0.f;
    if (t < NB * NCODE) aSum[t] = 0.f;
}

// ---------------- fused conv + distances + softmax + e accumulation ----------------
// 256 threads = 8 waves per WG; each WG processes 4 tiles of 64 pixels (256 WGs total)
__global__ __launch_bounds__(256) void fused_kernel(
    const float* __restrict__ x, const float* __restrict__ conv_b,
    const unsigned short* __restrict__ wb, const unsigned short* __restrict__ cb,
    const float* __restrict__ c2, const float* __restrict__ smooth,
    float* __restrict__ eAcc, float* __restrict__ aSum)
{
    __shared__ float          sStage[C_CH * SSTR];   // TDM staging: x tile f32 [channel][pixel]
    __shared__ unsigned short sX  [64 * XS];         // x tile  [pixel][channel] bf16
    __shared__ unsigned short sXp [64 * XS];         // xp tile [pixel][channel] bf16
    __shared__ unsigned short sXpT[C_CH * ATS];      // xp^T    [channel][pixel] bf16
    __shared__ float          sDist[64 * DSTR];      // logits / assign f32 [pixel][code]
    __shared__ unsigned short sAT [64 * ATS];        // assign^T [code][pixel] bf16
    __shared__ float          sX2[64];

    const int tid  = threadIdx.x;
    const int w    = tid >> 5;        // wave 0..7
    const int lane = tid & 31;
    const int row  = lane & 15;
    const int kh   = lane >> 4;
    const int wg   = blockIdx.x;      // 256 workgroups
    const int b    = wg >> 4;         // 16 WGs per batch
    const int tile0 = (wg & 15) * 4;  // each WG: 4 tiles of 64 pixels

    // stage-1 wave roles: pixel quarter / out-channel half
    const int pq    = w & 3;          // pixels 16*pq .. +15
    const int ohalf = w >> 2;         // o-tiles 8*ohalf .. +7
    // stage-2 wave roles: code tile / pixel-tile pair
    const int mt2    = w >> 1;        // codes 16*mt2 .. +15
    const int ntbase = 2 * (w & 1);   // pixel tiles ntbase .. +1

    v8f eacc[4][2];                   // stage-4: 4 code-tiles x 2 channel-tiles (ch 32w..32w+31)
#pragma unroll
    for (int mt = 0; mt < 4; ++mt)
#pragma unroll
        for (int j = 0; j < 2; ++j) eacc[mt][j] = (v8f){0,0,0,0,0,0,0,0};
    float aLocal = 0.f;

    const float* xb0 = x + (size_t)b * C_CH * HW;
    const unsigned stageOff = (unsigned)(uintptr_t)(void*)&sStage[0];

    // kick off DMA of first tile
    if (w == 0) tdm_load_tile(xb0 + tile0 * 64, stageOff);

    for (int it = 0; it < 4; ++it) {
        const int n0 = (tile0 + it) * 64;
        if (w == 0) __builtin_amdgcn_s_wait_tensorcnt(0);  // tile data landed in LDS
        __syncthreads();
        if (tid < 64) sX2[tid] = 0.f;

        // ---- convert staged f32 tile -> bf16, transposed [pixel][channel]
        for (int q = tid; q < C_CH * 16; q += 256) {
            int c = q >> 4, nq = q & 15;
            f32x4 v = *reinterpret_cast<const f32x4*>(&sStage[c * SSTR + nq * 4]);
            int nbase = nq * 4;
            sX[(nbase + 0) * XS + c] = f2bf_u(v.x);
            sX[(nbase + 1) * XS + c] = f2bf_u(v.y);
            sX[(nbase + 2) * XS + c] = f2bf_u(v.z);
            sX[(nbase + 3) * XS + c] = f2bf_u(v.w);
        }
        __syncthreads();
        // staging buffer free: start DMA of next tile, overlapped with WMMA stages below
        if (w == 0 && it + 1 < 4) tdm_load_tile(xb0 + n0 + 64, stageOff);

        // ---- stage 1: xp = x @ conv_w^T + conv_b  (wave: pixels 16pq..+15, o-tiles 8*ohalf..+7)
        float x2p[8];
#pragma unroll
        for (int r = 0; r < 8; ++r) x2p[r] = 0.f;
        const unsigned short* sxrow = sX + (16 * pq + row) * XS;
#pragma unroll 1
        for (int oi = 0; oi < 8; ++oi) {
            const int ot = 8 * ohalf + oi;
            v8f acc = (v8f){0,0,0,0,0,0,0,0};
            const unsigned short* wrow = wb + (16 * ot + row) * C_CH;
#pragma unroll
            for (int kt = 0; kt < 8; ++kt)
                acc = wmma_bf16(fragA(sxrow, kt, kh), fragB(wrow, kt, kh), acc);
            float bias = conv_b[16 * ot + row];
            union { u32x4 q; unsigned short s[8]; } pk;
#pragma unroll
            for (int r = 0; r < 8; ++r) {
                float v = acc[r] + bias;
                x2p[r] += v * v;
                unsigned short h = f2bf_u(v);
                pk.s[r] = h;                                   // xp^T: 8 consecutive pixels
                sXp[(16 * pq + r + 8 * kh) * XS + (16 * ot + row)] = h;
            }
            // one packed 16B store: xp^T[channel][pixel], pixels 16pq+8kh .. +7
            *reinterpret_cast<u32x4*>(&sXpT[(16 * ot + row) * ATS + 16 * pq + 8 * kh]) = pk.q;
        }
#pragma unroll
        for (int r = 0; r < 8; ++r) atomicAdd(&sX2[16 * pq + r + 8 * kh], x2p[r]);
        __syncthreads();

        // ---- stage 2: logits = -smooth*(x2 - 2 xp.c + c2)  (wave: codes 16*mt2..+15, 2 pixel tiles)
        const unsigned short* cbrow = cb + (16 * mt2 + row) * C_CH;
#pragma unroll
        for (int nt2 = 0; nt2 < 2; ++nt2) {
            const int nt = ntbase + nt2;
            v8f acc = (v8f){0,0,0,0,0,0,0,0};
            const unsigned short* xprow = sXp + (16 * nt + row) * XS;
#pragma unroll
            for (int kt = 0; kt < 8; ++kt)
                acc = wmma_bf16(fragA(cbrow, kt, kh), fragB(xprow, kt, kh), acc);
#pragma unroll
            for (int r = 0; r < 8; ++r) {
                int code = 16 * mt2 + 8 * kh + r;
                int pix  = 16 * nt + row;
                float d = sX2[pix] - 2.f * acc[r] + c2[code];
                sDist[pix * DSTR + code] = -smooth[code] * d;
            }
        }
        __syncthreads();

        // ---- stage 3: softmax over 64 codes (one pixel per thread)
        if (tid < 64) {
            int n = tid;
            float m = -1e30f;
            for (int k = 0; k < NCODE; ++k) m = fmaxf(m, sDist[n * DSTR + k]);
            float s = 0.f;
            for (int k = 0; k < NCODE; ++k) {
                float e = __expf(sDist[n * DSTR + k] - m);
                sDist[n * DSTR + k] = e; s += e;
            }
            float inv = 1.f / s;
            for (int k = 0; k < NCODE; ++k) {
                float a = sDist[n * DSTR + k] * inv;
                sDist[n * DSTR + k] = a;
                sAT[k * ATS + n] = f2bf_u(a);
            }
        }
        __syncthreads();
        if (tid < 64) {      // a_sum column reduction (kept in registers across tiles)
            float s = 0.f;
            for (int n = 0; n < 64; ++n) s += sDist[n * DSTR + tid];
            aLocal += s;
        }

        // ---- stage 4: e += assign^T @ xp  (wave: channel tiles 2w..2w+1), K = 64 pixels
        //      A from sAT rows (codes), B from sXpT rows (channels) — all b128 LDS loads
#pragma unroll
        for (int mt = 0; mt < 4; ++mt) {
            const unsigned short* atrow = sAT + (16 * mt + row) * ATS;
#pragma unroll
            for (int j = 0; j < 2; ++j) {
                const unsigned short* xtrow = sXpT + (16 * (2 * w + j) + row) * ATS;
#pragma unroll
                for (int kt = 0; kt < 2; ++kt)
                    eacc[mt][j] = wmma_bf16(fragA(atrow, kt, kh),
                                            fragB(xtrow, kt, kh), eacc[mt][j]);
            }
        }
        __syncthreads();
    }

    // ---- flush partial e and a_sum
    float* eb = eAcc + (size_t)b * NCODE * C_CH;
#pragma unroll
    for (int mt = 0; mt < 4; ++mt)
#pragma unroll
        for (int j = 0; j < 2; ++j)
#pragma unroll
            for (int r = 0; r < 8; ++r) {
                int code = 16 * mt + 8 * kh + r;
                int cch  = 16 * (2 * w + j) + row;
                atomicAdd(&eb[code * C_CH + cch], eacc[mt][j][r]);
            }
    if (tid < 64) atomicAdd(&aSum[b * NCODE + tid], aLocal);
}

// ---------------- impact = sigmoid(fc(mean_k(e - a_sum*codebook))) ----------------
__global__ __launch_bounds__(256) void impact_kernel(
    const float* __restrict__ eAcc, const float* __restrict__ aSum,
    const float* __restrict__ codebook, const float* __restrict__ fc_w,
    const float* __restrict__ fc_b, float* __restrict__ impact)
{
    __shared__ float eavg[C_CH];
    int b = blockIdx.x, c = threadIdx.x;
    float s = 0.f;
    for (int k = 0; k < NCODE; ++k)
        s += eAcc[((size_t)b * NCODE + k) * C_CH + c] - aSum[b * NCODE + k] * codebook[k * C_CH + c];
    eavg[c] = s * (1.f / (float)NCODE);
    __syncthreads();
    float d = fc_b[c];
    for (int cc = 0; cc < C_CH; ++cc) d += eavg[cc] * fc_w[c * C_CH + cc];
    impact[b * C_CH + c] = 1.f / (1.f + __expf(-d));
}

// ---------------- out = x * (1 + impact[b,c]) ----------------
__global__ __launch_bounds__(256) void scale_kernel(
    const float* __restrict__ x, const float* __restrict__ impact, float* __restrict__ out)
{
    int bc = blockIdx.x;                    // B*C blocks
    float s = 1.f + impact[bc];
    const f32x4* xi = reinterpret_cast<const f32x4*>(x) + (size_t)bc * (HW / 4);
    f32x4* oo = reinterpret_cast<f32x4*>(out) + (size_t)bc * (HW / 4);
    for (int i = threadIdx.x; i < HW / 4; i += 256) {
        f32x4 v = xi[i];
        v.x *= s; v.y *= s; v.z *= s; v.w *= s;
        oo[i] = v;
    }
}

extern "C" void kernel_launch(void* const* d_in, const int* in_sizes, int n_in,
                              void* d_out, int out_size, void* d_ws, size_t ws_size,
                              hipStream_t stream) {
    const float* x          = (const float*)d_in[0];
    const float* conv_w     = (const float*)d_in[1];
    const float* conv_b     = (const float*)d_in[2];
    const float* codebook   = (const float*)d_in[3];
    const float* log_smooth = (const float*)d_in[4];
    const float* fc_w       = (const float*)d_in[5];
    const float* fc_b       = (const float*)d_in[6];
    float* out = (float*)d_out;

    char* ws = (char*)d_ws;
    unsigned short* wb  = (unsigned short*)(ws + 0);        // 131072 B
    unsigned short* cb  = (unsigned short*)(ws + 131072);   //  32768 B
    float* c2     = (float*)(ws + 163840);                  //    256 B
    float* smooth = (float*)(ws + 164096);                  //    256 B
    float* aSum   = (float*)(ws + 164352);                  //   4096 B
    float* impact = (float*)(ws + 168448);                  //  16384 B
    float* eAcc   = (float*)(ws + 184832);                  // 1048576 B

    prep_kernel<<<1024, 256, 0, stream>>>(conv_w, codebook, log_smooth,
                                          wb, cb, c2, smooth, eAcc, aSum);
    fused_kernel<<<256, 256, 0, stream>>>(x, conv_b, wb, cb, c2, smooth, eAcc, aSum);
    impact_kernel<<<NB, 256, 0, stream>>>(eAcc, aSum, codebook, fc_w, fc_b, impact);
    scale_kernel<<<NB * C_CH, 256, 0, stream>>>(x, impact, out);
}
